// MLPFlowMatchingModel_37177236914856
// MI455X (gfx1250) — compile-verified, matrix operand order checked by reading, not run on hardware
//
#include <hip/hip_runtime.h>
#include <hip/hip_bf16.h>

#define VN   50000
#define EN   800000
#define HD   128
#define LATN 64
#define TDN  16
#define LAYERS 4
#define EPSN 1e-5f

typedef __attribute__((ext_vector_type(16))) _Float16 v16h;
typedef __attribute__((ext_vector_type(8)))  float    v8f;

union FragU { v16h v; _Float16 h[16]; float4 q[2]; };

__device__ __forceinline__ float siluf(float x) { return x / (1.0f + __expf(-x)); }

// Load one 16x32 f16 fragment (A from LDS, or B from transposed [N][Kpad] weights).
// Lane n<16: rows/cols n, K = kbase+{0..7, 16..23}; lane>=16: K = kbase+{8..15, 24..31}.
__device__ __forceinline__ v16h load_frag(const _Float16* base, int strideH, int kbase) {
  int lane = threadIdx.x & 31;
  int rc   = lane & 15;
  int k0   = kbase + ((lane & 16) ? 8 : 0);
  FragU f;
  const _Float16* p = base + rc * strideH + k0;
  f.q[0] = *(const float4*)(p);
  f.q[1] = *(const float4*)(p + 16);
  return f.v;
}

__device__ __forceinline__ v8f wmma_f16(v16h a, v16h b, v8f c) {
  return __builtin_amdgcn_wmma_f32_16x16x32_f16(false, a, false, b, (short)0, c, false, false);
}

// ---------------- small prep kernels ----------------

// w: [layers][K][N] f32  ->  wt: [layers][N][Kpad] f16 (zero padded)
__global__ void k_transpose_w(const float* __restrict__ w, _Float16* __restrict__ wt,
                              int K, int Kpad, int N, int layers) {
  int idx = blockIdx.x * blockDim.x + threadIdx.x;
  int total = layers * N * Kpad;
  if (idx >= total) return;
  int kp = idx % Kpad;
  int n  = (idx / Kpad) % N;
  int l  = idx / (Kpad * N);
  float val = (kp < K) ? w[((size_t)l * K + kp) * N + n] : 0.0f;
  wt[idx] = (_Float16)val;
}

__global__ void k_temb(const float* __restrict__ t,
                       const float* __restrict__ w1, const float* __restrict__ b1,
                       const float* __restrict__ w2, const float* __restrict__ b2,
                       float* __restrict__ t_emb) {
  __shared__ float hid[TDN];
  int j = threadIdx.x;                // 16 threads
  float ts = t[0];
  float x = ts * w1[j] + b1[j];
  hid[j] = siluf(x);
  __syncthreads();
  float acc = b2[j];
  for (int k = 0; k < TDN; ++k) acc += hid[k] * w2[k * TDN + j];
  t_emb[j] = acc;
}

// h0 = [z[batch[v]], t_emb] @ cp_w + cp_b ; also x_t = (1-t)*pos0 + t*pos1
__global__ __launch_bounds__(HD) void k_h0(const float* __restrict__ pos0,
                                           const float* __restrict__ pos1,
                                           const float* __restrict__ z,
                                           const float* __restrict__ t,
                                           const int* __restrict__ batch,
                                           const float* __restrict__ t_emb,
                                           const float* __restrict__ cp_w,
                                           const float* __restrict__ cp_b,
                                           float* __restrict__ x_t,
                                           float* __restrict__ h32,
                                           _Float16* __restrict__ h16) {
  __shared__ float in80[LATN + TDN];
  int v = blockIdx.x;
  int c = threadIdx.x;
  float ts = t[0];
  if (c < LATN)            in80[c] = z[(size_t)batch[v] * LATN + c];
  else if (c < LATN + TDN) in80[c] = t_emb[c - LATN];
  if (c < 3) x_t[v * 3 + c] = (1.0f - ts) * pos0[v * 3 + c] + ts * pos1[v * 3 + c];
  __syncthreads();
  float acc = cp_b[c];
  for (int k = 0; k < LATN + TDN; ++k) acc += in80[k] * cp_w[k * HD + c];
  h32[(size_t)v * HD + c] = acc;
  h16[(size_t)v * HD + c] = (_Float16)acc;
}

__global__ void k_zero(float* __restrict__ p, int n) {
  int i = blockIdx.x * blockDim.x + threadIdx.x;
  if (i < n) p[i] = 0.0f;
}

// ---------------- edge message + aggregate (per layer) ----------------
// 32 edges per block, 8 waves; wave w owns output columns [16w,16w+16).
__global__ __launch_bounds__(256) void k_edge(const _Float16* __restrict__ h16,
                                              const float* __restrict__ x_t,
                                              const int* __restrict__ eidx,
                                              const _Float16* __restrict__ w1t,
                                              const float* __restrict__ b1,
                                              const _Float16* __restrict__ w2t,
                                              const float* __restrict__ b2,
                                              float* __restrict__ agg) {
  __shared__ __align__(16) _Float16 A1[32][288];
  __shared__ __align__(16) _Float16 A2[32][128];
  __shared__ int sdst[32];

  const int t  = threadIdx.x;
  const int e0 = blockIdx.x * 32;
  const int r  = t >> 3, c = t & 7;            // 8 threads per edge row
  const int e  = e0 + r;
  const int s  = eidx[e];                      // src
  const int d  = eidx[EN + e];                 // dst
  if (c == 0) sdst[r] = d;

  // gather h16[dst] -> A1[r][0:128], h16[src] -> A1[r][128:256]  (16B loads)
  const float4* pd = (const float4*)(h16 + (size_t)d * HD);
  const float4* ps = (const float4*)(h16 + (size_t)s * HD);
  float4* a1row = (float4*)(&A1[r][0]);
  a1row[c * 2]          = pd[c * 2];
  a1row[c * 2 + 1]      = pd[c * 2 + 1];
  a1row[16 + c * 2]     = ps[c * 2];
  a1row[16 + c * 2 + 1] = ps[c * 2 + 1];
  if (c == 0) {
    A1[r][256] = (_Float16)(x_t[d * 3 + 0] - x_t[s * 3 + 0]);
    A1[r][257] = (_Float16)(x_t[d * 3 + 1] - x_t[s * 3 + 1]);
    A1[r][258] = (_Float16)(x_t[d * 3 + 2] - x_t[s * 3 + 2]);
    for (int k = 259; k < 288; ++k) A1[r][k] = (_Float16)0.0f;
  }
  __syncthreads();

  const int w    = t >> 5;
  const int lane = t & 31;
  const int ncol = w * 16;
  const int gcol = ncol + (lane & 15);
  const int rowoff = (lane & 16) ? 8 : 0;

  // GEMM1: [32 x 288] x [288 x 16-per-wave]
  v8f acc0 = {}, acc1 = {};
  const _Float16* wb1 = w1t + (size_t)ncol * 288;
  #pragma unroll
  for (int kt = 0; kt < 9; ++kt) {
    v16h bf = load_frag(wb1, 288, kt * 32);
    v16h a0 = load_frag(&A1[0][0],  288, kt * 32);
    v16h a1 = load_frag(&A1[16][0], 288, kt * 32);
    acc0 = wmma_f16(a0, bf, acc0);
    acc1 = wmma_f16(a1, bf, acc1);
  }
  float bias1 = b1[gcol];
  #pragma unroll
  for (int rr = 0; rr < 8; ++rr) {
    int row = rr + rowoff;
    A2[row][gcol]      = (_Float16)siluf(acc0[rr] + bias1);
    A2[row + 16][gcol] = (_Float16)siluf(acc1[rr] + bias1);
  }
  __syncthreads();

  // GEMM2: [32 x 128] x [128 x 16-per-wave]
  v8f acc2 = {}, acc3 = {};
  const _Float16* wb2 = w2t + (size_t)ncol * 128;
  #pragma unroll
  for (int kt = 0; kt < 4; ++kt) {
    v16h bf = load_frag(wb2, 128, kt * 32);
    v16h a0 = load_frag(&A2[0][0],  128, kt * 32);
    v16h a1 = load_frag(&A2[16][0], 128, kt * 32);
    acc2 = wmma_f16(a0, bf, acc2);
    acc3 = wmma_f16(a1, bf, acc3);
  }
  float bias2 = b2[gcol];
  #pragma unroll
  for (int rr = 0; rr < 8; ++rr) {
    int row = rr + rowoff;
    float m0 = siluf(acc2[rr] + bias2);
    float m1 = siluf(acc3[rr] + bias2);
    atomicAdd(&agg[(size_t)sdst[row] * HD + gcol], m0);
    atomicAdd(&agg[(size_t)sdst[row + 16] * HD + gcol], m1);
  }
}

// ---------------- node update + LayerNorm (per layer, in place) ----------------
__global__ __launch_bounds__(256) void k_node(const _Float16* h16_in,
                                              const float* h32_in,
                                              const float* __restrict__ agg,
                                              const _Float16* __restrict__ w1t,
                                              const float* __restrict__ b1,
                                              const _Float16* __restrict__ w2t,
                                              const float* __restrict__ b2,
                                              const float* __restrict__ lng,
                                              const float* __restrict__ lnb,
                                              float* h32_out,
                                              _Float16* h16_out) {
  __shared__ __align__(16) _Float16 A1[32][256];
  __shared__ __align__(16) _Float16 A2[32][128];
  __shared__ __align__(16) float    HS[32][128];
  __shared__ float MU[32], RS[32];

  const int t  = threadIdx.x;
  const int v0 = blockIdx.x * 32;
  const int r  = t >> 3, c = t & 7;
  const int v  = min(v0 + r, VN - 1);

  // stage concat(h, agg) as f16
  const float4* ph = (const float4*)(h16_in + (size_t)v * HD);
  float4* a1row = (float4*)(&A1[r][0]);
  a1row[c * 2]     = ph[c * 2];
  a1row[c * 2 + 1] = ph[c * 2 + 1];
  const float* pa = agg + (size_t)v * HD;
  #pragma unroll
  for (int i = 0; i < 16; ++i) A1[r][128 + c * 16 + i] = (_Float16)pa[c * 16 + i];
  __syncthreads();

  const int w    = t >> 5;
  const int lane = t & 31;
  const int ncol = w * 16;
  const int gcol = ncol + (lane & 15);
  const int rowoff = (lane & 16) ? 8 : 0;

  v8f acc0 = {}, acc1 = {};
  const _Float16* wb1 = w1t + (size_t)ncol * 256;
  #pragma unroll
  for (int kt = 0; kt < 8; ++kt) {
    v16h bf = load_frag(wb1, 256, kt * 32);
    v16h a0 = load_frag(&A1[0][0],  256, kt * 32);
    v16h a1 = load_frag(&A1[16][0], 256, kt * 32);
    acc0 = wmma_f16(a0, bf, acc0);
    acc1 = wmma_f16(a1, bf, acc1);
  }
  float bias1 = b1[gcol];
  #pragma unroll
  for (int rr = 0; rr < 8; ++rr) {
    int row = rr + rowoff;
    A2[row][gcol]      = (_Float16)siluf(acc0[rr] + bias1);
    A2[row + 16][gcol] = (_Float16)siluf(acc1[rr] + bias1);
  }
  __syncthreads();

  v8f acc2 = {}, acc3 = {};
  const _Float16* wb2 = w2t + (size_t)ncol * 128;
  #pragma unroll
  for (int kt = 0; kt < 4; ++kt) {
    v16h bf = load_frag(wb2, 128, kt * 32);
    v16h a0 = load_frag(&A2[0][0],  128, kt * 32);
    v16h a1 = load_frag(&A2[16][0], 128, kt * 32);
    acc2 = wmma_f16(a0, bf, acc2);
    acc3 = wmma_f16(a1, bf, acc3);
  }
  float bias2 = b2[gcol];
  #pragma unroll
  for (int rr = 0; rr < 8; ++rr) {
    int row = rr + rowoff;
    int vA = min(v0 + row, VN - 1);
    int vB = min(v0 + row + 16, VN - 1);
    HS[row][gcol]      = h32_in[(size_t)vA * HD + gcol] + acc2[rr] + bias2;  // residual
    HS[row + 16][gcol] = h32_in[(size_t)vB * HD + gcol] + acc3[rr] + bias2;
  }
  __syncthreads();

  if (t < 32) {
    float mu = 0.0f;
    for (int k = 0; k < HD; ++k) mu += HS[t][k];
    mu *= (1.0f / HD);
    float var = 0.0f;
    for (int k = 0; k < HD; ++k) { float dd = HS[t][k] - mu; var += dd * dd; }
    var *= (1.0f / HD);
    MU[t] = mu;
    RS[t] = rsqrtf(var + EPSN);
  }
  __syncthreads();

  int row = t >> 3, cc = (t & 7) * 16;
  int vv = v0 + row;
  if (vv < VN) {
    float mu = MU[row], rs = RS[row];
    #pragma unroll
    for (int i = 0; i < 16; ++i) {
      int gc = cc + i;
      float y = lng[gc] * (HS[row][gc] - mu) * rs + lnb[gc];
      h32_out[(size_t)vv * HD + gc] = y;
      h16_out[(size_t)vv * HD + gc] = (_Float16)y;
    }
  }
}

// ---------------- out-proj + MSE reduce ----------------
__global__ __launch_bounds__(256) void k_loss(const float* __restrict__ h32,
                                              const float* __restrict__ pos0,
                                              const float* __restrict__ pos1,
                                              const float* __restrict__ opw,
                                              const float* __restrict__ opb,
                                              float* __restrict__ out) {
  __shared__ float red[256];
  float local = 0.0f;
  for (int v = blockIdx.x * 256 + threadIdx.x; v < VN; v += gridDim.x * 256) {
    const float* hr = h32 + (size_t)v * HD;
    float a0 = opb[0], a1 = opb[1], a2 = opb[2];
    for (int k = 0; k < HD; ++k) {
      float hv = hr[k];
      a0 += hv * opw[k * 3 + 0];
      a1 += hv * opw[k * 3 + 1];
      a2 += hv * opw[k * 3 + 2];
    }
    float d0 = a0 - (pos1[v * 3 + 0] - pos0[v * 3 + 0]);
    float d1 = a1 - (pos1[v * 3 + 1] - pos0[v * 3 + 1]);
    float d2 = a2 - (pos1[v * 3 + 2] - pos0[v * 3 + 2]);
    local += d0 * d0 + d1 * d1 + d2 * d2;
  }
  red[threadIdx.x] = local;
  __syncthreads();
  for (int s = 128; s > 0; s >>= 1) {
    if (threadIdx.x < s) red[threadIdx.x] += red[threadIdx.x + s];
    __syncthreads();
  }
  if (threadIdx.x == 0) atomicAdd(out, red[0] * (1.0f / (float)(VN * 3)));
}

// ---------------- launch ----------------
extern "C" void kernel_launch(void* const* d_in, const int* in_sizes, int n_in,
                              void* d_out, int out_size, void* d_ws, size_t ws_size,
                              hipStream_t stream) {
  const float* pos0  = (const float*)d_in[0];
  const float* pos1  = (const float*)d_in[1];
  const float* z     = (const float*)d_in[2];
  const float* t     = (const float*)d_in[3];
  const int*   eidx  = (const int*)d_in[4];
  const int*   batch = (const int*)d_in[5];
  const float* te_w1 = (const float*)d_in[6];
  const float* te_b1 = (const float*)d_in[7];
  const float* te_w2 = (const float*)d_in[8];
  const float* te_b2 = (const float*)d_in[9];
  const float* cp_w  = (const float*)d_in[10];
  const float* cp_b  = (const float*)d_in[11];
  const float* ew1   = (const float*)d_in[12];
  const float* eb1   = (const float*)d_in[13];
  const float* ew2   = (const float*)d_in[14];
  const float* eb2   = (const float*)d_in[15];
  const float* nw1   = (const float*)d_in[16];
  const float* nb1   = (const float*)d_in[17];
  const float* nw2   = (const float*)d_in[18];
  const float* nb2   = (const float*)d_in[19];
  const float* ln_g  = (const float*)d_in[20];
  const float* ln_b  = (const float*)d_in[21];
  const float* op_w  = (const float*)d_in[22];
  const float* op_b  = (const float*)d_in[23];

  char* ws = (char*)d_ws;
  size_t off = 0;
  auto alloc = [&](size_t bytes) {
    size_t o = off;
    off = (off + bytes + 255) & ~(size_t)255;
    return o;
  };
  _Float16* h16   = (_Float16*)(ws + alloc((size_t)VN * HD * 2));
  float*    h32   = (float*)   (ws + alloc((size_t)VN * HD * 4));
  float*    agg   = (float*)   (ws + alloc((size_t)VN * HD * 4));
  float*    x_t   = (float*)   (ws + alloc((size_t)VN * 3 * 4));
  float*    t_emb = (float*)   (ws + alloc((size_t)TDN * 4));
  _Float16* ew1t  = (_Float16*)(ws + alloc((size_t)LAYERS * HD * 288 * 2));
  _Float16* ew2t  = (_Float16*)(ws + alloc((size_t)LAYERS * HD * 128 * 2));
  _Float16* nw1t  = (_Float16*)(ws + alloc((size_t)LAYERS * HD * 256 * 2));
  _Float16* nw2t  = (_Float16*)(ws + alloc((size_t)LAYERS * HD * 128 * 2));

  int n1 = LAYERS * HD * 288;
  k_transpose_w<<<(n1 + 255) / 256, 256, 0, stream>>>(ew1, ew1t, 259, 288, HD, LAYERS);
  int n2 = LAYERS * HD * 128;
  k_transpose_w<<<(n2 + 255) / 256, 256, 0, stream>>>(ew2, ew2t, 128, 128, HD, LAYERS);
  int n3 = LAYERS * HD * 256;
  k_transpose_w<<<(n3 + 255) / 256, 256, 0, stream>>>(nw1, nw1t, 256, 256, HD, LAYERS);
  k_transpose_w<<<(n2 + 255) / 256, 256, 0, stream>>>(nw2, nw2t, 128, 128, HD, LAYERS);

  k_temb<<<1, TDN, 0, stream>>>(t, te_w1, te_b1, te_w2, te_b2, t_emb);
  k_h0<<<VN, HD, 0, stream>>>(pos0, pos1, z, t, batch, t_emb, cp_w, cp_b, x_t, h32, h16);

  for (int l = 0; l < LAYERS; ++l) {
    k_zero<<<(VN * HD + 255) / 256, 256, 0, stream>>>(agg, VN * HD);
    k_edge<<<EN / 32, 256, 0, stream>>>(h16, x_t, eidx,
                                        ew1t + (size_t)l * HD * 288, eb1 + l * HD,
                                        ew2t + (size_t)l * HD * 128, eb2 + l * HD,
                                        agg);
    k_node<<<(VN + 31) / 32, 256, 0, stream>>>(h16, h32, agg,
                                               nw1t + (size_t)l * HD * 256, nb1 + l * HD,
                                               nw2t + (size_t)l * HD * 128, nb2 + l * HD,
                                               ln_g + l * HD, ln_b + l * HD,
                                               h32, h16);
  }

  k_zero<<<1, 256, 0, stream>>>((float*)d_out, out_size);
  k_loss<<<256, 256, 0, stream>>>(h32, pos0, pos1, op_w, op_b, (float*)d_out);
}
